// Mlp_VAE_11029476016537
// MI455X (gfx1250) — matrix operand order repaired and checked by reading, not compiled, not run on hardware
//
#include <hip/hip_runtime.h>
#include <hip/hip_bf16.h>
#include <math.h>

// ---------------- problem constants ----------------
#define NB      65536          // batch rows
#define NL      512            // L == H == K (all 512)
#define NE      1024           // expander group
#define GRP     (NE * NL)      // 524288 = 2^19 elements per LN group
#define NGROUP  (NB / NE)      // 64 groups
#define BLTOT   ((size_t)NB * NL)   // 33,554,432 elements per activation
#define WSZ     ((size_t)NL * NL)   // 262144 = 2^18 (one weight matrix)

typedef __attribute__((ext_vector_type(16))) __bf16 v16bf;
typedef __attribute__((ext_vector_type(2)))  __bf16 bfx2;
typedef __attribute__((ext_vector_type(8)))  float  v8f;
typedef __attribute__((ext_vector_type(8)))  unsigned int v8u;

// hardware f32 -> bf16 (RNE) via fptrunc
__device__ __forceinline__ unsigned short f2bf(float f) {
  __bf16 h = (__bf16)f;
  return __builtin_bit_cast(unsigned short, h);
}
__device__ __forceinline__ unsigned int pk2bf(float a, float b) {
  bfx2 t; t.x = (__bf16)a; t.y = (__bf16)b;
  return __builtin_bit_cast(unsigned int, t);
}

__device__ __forceinline__ float gelu_exact(float v) {
  return 0.5f * v * (1.0f + erff(v * 0.7071067811865476f));
}

// gfx1250 async copy: global (16B) -> LDS, tracked by ASYNCcnt
__device__ __forceinline__ void async_b128(unsigned lds_off, const void* gaddr) {
  asm volatile("global_load_async_to_lds_b128 %0, %1, off"
               :: "v"(lds_off), "v"(gaddr) : "memory");
}

// ---------------- WMMA GEMM: C[M x 512] = A[M x 512] @ W[512 x 512] ----------------
// A: bf16 row-major. Wt: bf16 PRE-TRANSPOSED [n][k].
// MODE 0: outb = bf16(gelu(acc + bias[n]))
// MODE 1: outf = acc + bias[n] (+ res[m][n] if res != null)
// MODE 2: outf = bias[n] - 2*acc      (VQ score; bias = ||cb_j||^2)
constexpr int BM = 128, BN = 128, BK = 32;
constexpr int ASTR = 40;  // padded row stride (bf16): 80B = 5*16B -> 16B aligned rows
constexpr int KSTEPS = NL / BK;      // 16

template <int MODE>
__global__ __launch_bounds__(256)
void gemm512(const unsigned short* __restrict__ A, const unsigned short* __restrict__ Wt,
             const float* __restrict__ bias, const float* __restrict__ res,
             float* __restrict__ outf, unsigned short* __restrict__ outb) {
  __shared__ unsigned short As[2][BM * ASTR];  // [buf][row][k]
  __shared__ unsigned short Bs[2][BN * ASTR];  // [buf][n][k]

  const int tid  = threadIdx.x;
  const int lane = tid & 31;
  const int wave = tid >> 5;          // 8 waves: 4 (M) x 2 (N)
  const int wm   = wave >> 1;         // 0..3 -> 32-row slab
  const int wn   = wave & 1;          // 0..1 -> 64-col slab
  const int half = lane >> 4;
  const int l15  = lane & 15;

  const int m0 = blockIdx.y * BM;
  const int n0 = blockIdx.x * BN;

  // staging assignment: 2 x b128 per thread for A, 2 for B (4 async ops / tile)
  const int sr = tid >> 2;            // 0..63
  const int sc = (tid & 3) * 8;       // 0,8,16,24 (bf16 elems)

  v8f acc[2][4] = {};

  auto stage = [&](int k0, int b) {
#pragma unroll
    for (int i = 0; i < 2; ++i) {
      const int r = sr + i * 64;      // 0..127
      async_b128((unsigned)(uintptr_t)(&As[b][r * ASTR + sc]),
                 A + (size_t)(m0 + r) * NL + k0 + sc);
    }
#pragma unroll
    for (int i = 0; i < 2; ++i) {
      const int r = sr + i * 64;
      async_b128((unsigned)(uintptr_t)(&Bs[b][r * ASTR + sc]),
                 Wt + (size_t)(n0 + r) * NL + k0 + sc);
    }
  };

  stage(0, 0);                         // prologue: tile 0 in flight
  for (int it = 0; it < KSTEPS; ++it) {
    const int buf = it & 1;
    if (it + 1 < KSTEPS) {
      stage((it + 1) * BK, buf ^ 1);   // keep next tile in flight
      __builtin_prefetch(A + (size_t)(m0 + sr + 64) * NL + (it + 1) * BK + sc, 0, 3);
      asm volatile("s_wait_asynccnt 4" ::: "memory");   // drain tile 'it' only
    } else {
      asm volatile("s_wait_asynccnt 0" ::: "memory");
    }
    __syncthreads();

    // ---- load fragments (ds_load_b128 x12), then 2x4 WMMA nest ----
    v16bf af[2];
#pragma unroll
    for (int mi = 0; mi < 2; ++mi) {
      // A frag 16x32: e<8 -> k=half*8+e ; e>=8 -> k=16+half*8+(e-8)
      const int row = wm * 32 + mi * 16 + l15;
      uint4 x0 = *(const uint4*)(&As[buf][row * ASTR + half * 8]);
      uint4 x1 = *(const uint4*)(&As[buf][row * ASTR + 16 + half * 8]);
      v8u au = { x0.x, x0.y, x0.z, x0.w, x1.x, x1.y, x1.z, x1.w };
      af[mi] = __builtin_bit_cast(v16bf, au);
    }
    v16bf bfr[4];
#pragma unroll
    for (int ni = 0; ni < 4; ++ni) {
      // B frag 32x16: lane -> col, e<8 -> k=half*8+e ; e>=8 -> k=16+half*8+(e-8)
      const int col = wn * 64 + ni * 16 + l15;
      uint4 y0 = *(const uint4*)(&Bs[buf][col * ASTR + half * 8]);
      uint4 y1 = *(const uint4*)(&Bs[buf][col * ASTR + 16 + half * 8]);
      v8u bu = { y0.x, y0.y, y0.z, y0.w, y1.x, y1.y, y1.z, y1.w };
      bfr[ni] = __builtin_bit_cast(v16bf, bu);
    }
#pragma unroll
    for (int mi = 0; mi < 2; ++mi)
#pragma unroll
      for (int ni = 0; ni < 4; ++ni)
        acc[mi][ni] = __builtin_amdgcn_wmma_f32_16x16x32_bf16(
            false, af[mi], false, bfr[ni], (short)0, acc[mi][ni], false, false);
    __syncthreads();
  }

  // ---- epilogue: D layout: vgpr e, lanes0-15 -> m=e, lanes16-31 -> m=e+8; n=l15 ----
#pragma unroll
  for (int mi = 0; mi < 2; ++mi) {
#pragma unroll
    for (int ni = 0; ni < 4; ++ni) {
      const int n  = n0 + wn * 64 + ni * 16 + l15;
      const float bn = bias[n];
#pragma unroll
      for (int e = 0; e < 8; ++e) {
        const int m = m0 + wm * 32 + mi * 16 + e + 8 * half;
        float v = acc[mi][ni][e];
        if (MODE == 0) {
          outb[(size_t)m * NL + n] = f2bf(gelu_exact(v + bn));
        } else if (MODE == 1) {
          v += bn;
          if (res) v += res[(size_t)m * NL + n];
          outf[(size_t)m * NL + n] = v;
        } else {
          outf[(size_t)m * NL + n] = bn - 2.0f * v;
        }
      }
    }
  }
}

// ---------------- converters ----------------
__global__ __launch_bounds__(256)
void conv_plain(const float* __restrict__ in, unsigned short* __restrict__ out, size_t n) {
  const size_t i = ((size_t)blockIdx.x * 256 + threadIdx.x) * 4;
  if (i >= n) return;
  float4 v = *(const float4*)(in + i);
  *(uint2*)(out + i) = make_uint2(pk2bf(v.x, v.y), pk2bf(v.z, v.w));
}

// out[s][n][k] = bf16(in[s][k][n])   (pre-transpose weights / codebook)
__global__ __launch_bounds__(256)
void conv_transpose(const float* __restrict__ in, unsigned short* __restrict__ out, int nsl) {
  const size_t i = (size_t)blockIdx.x * 256 + threadIdx.x;
  if (i >= (size_t)nsl * WSZ) return;
  const size_t s = i >> 18;               // WSZ = 2^18
  const size_t r = i & (WSZ - 1);
  const int k = (int)(r >> 9), n = (int)(r & (NL - 1));
  out[s * WSZ + (size_t)n * NL + k] = f2bf(in[i]);
}

// ---------------- RecoverLN: deterministic 3-stage over 2^19-elem groups ----------------
__global__ __launch_bounds__(256)
void rln_partial(const float* __restrict__ in, float* __restrict__ psum,
                 float* __restrict__ psq) {
  const int g = blockIdx.x, blk = blockIdx.y;       // 64 x 32
  const float4* p4 = (const float4*)(in + (size_t)g * GRP + (size_t)blk * (GRP / 32));
  float s = 0.f, q = 0.f;
  for (int i = threadIdx.x; i < (GRP / 32) / 4; i += 256) {
    float4 v = p4[i];
    s += v.x + v.y + v.z + v.w;
    q += v.x * v.x + v.y * v.y + v.z * v.z + v.w * v.w;
  }
  __shared__ float sh[256], sh2[256];
  sh[threadIdx.x] = s; sh2[threadIdx.x] = q;
  __syncthreads();
  for (int st = 128; st > 0; st >>= 1) {
    if (threadIdx.x < st) { sh[threadIdx.x] += sh[threadIdx.x + st]; sh2[threadIdx.x] += sh2[threadIdx.x + st]; }
    __syncthreads();
  }
  if (threadIdx.x == 0) { psum[g * 32 + blk] = sh[0]; psq[g * 32 + blk] = sh2[0]; }
}

__global__ void rln_finalize(const float* __restrict__ psum, const float* __restrict__ psq,
                             float* __restrict__ mu, float* __restrict__ rstd) {
  const int g = threadIdx.x;
  if (g >= NGROUP) return;
  float s = 0.f, q = 0.f;
  for (int i = 0; i < 32; ++i) { s += psum[g * 32 + i]; q += psq[g * 32 + i]; }
  const float m = s * (1.0f / (float)GRP);
  const float v = q * (1.0f / (float)GRP) - m * m;
  mu[g] = m;
  rstd[g] = rsqrtf(v + 1e-5f);
}

__global__ __launch_bounds__(256)
void rln_apply(const float* __restrict__ in, const float* __restrict__ gamma,
               const float* __restrict__ beta, const float* __restrict__ mu,
               const float* __restrict__ rstd,
               float* __restrict__ of32, unsigned short* __restrict__ obf) {
  const size_t i = ((size_t)blockIdx.x * 256 + threadIdx.x) * 4;
  if (i >= BLTOT) return;
  const int    g = (int)(i >> 19);
  const size_t p = i & (GRP - 1);
  const float  m = mu[g], r = rstd[g];
  float4 v  = *(const float4*)(in + i);
  float4 gm = *(const float4*)(gamma + p);
  float4 bt = *(const float4*)(beta + p);
  float4 o;
  o.x = (v.x - m) * r * gm.x + bt.x;
  o.y = (v.y - m) * r * gm.y + bt.y;
  o.z = (v.z - m) * r * gm.z + bt.z;
  o.w = (v.w - m) * r * gm.w + bt.w;
  if (of32) *(float4*)(of32 + i) = o;
  if (obf)  *(uint2*)(obf + i) = make_uint2(pk2bf(o.x, o.y), pk2bf(o.z, o.w));
}

// ---------------- VQ pieces ----------------
__global__ void cb_norm(const float* __restrict__ cb, float* __restrict__ cbn) {
  const int j = blockIdx.x * 256 + threadIdx.x;
  if (j >= NL) return;
  float s = 0.f;
  for (int l = 0; l < NL; ++l) { float v = cb[(size_t)l * NL + j]; s += v * v; }
  cbn[j] = s;
}

__global__ __launch_bounds__(256)
void argmin_rows(const float* __restrict__ d, int* __restrict__ amin,
                 float* __restrict__ aminf) {
  const int row  = blockIdx.x * 8 + (threadIdx.x >> 5);
  const int lane = threadIdx.x & 31;
  const float* p = d + (size_t)row * NL;
  float best = 3.402823466e38f;
  int   bi   = 0;
  for (int j = lane; j < NL; j += 32) {
    float v = p[j];
    if (v < best) { best = v; bi = j; }
  }
  for (int off = 16; off > 0; off >>= 1) {
    float ov = __shfl_down(best, off, 32);
    int   oi = __shfl_down(bi,   off, 32);
    if (ov < best || (ov == best && oi < bi)) { best = ov; bi = oi; }
  }
  if (lane == 0) { amin[row] = bi; aminf[row] = (float)bi; }
}

__global__ __launch_bounds__(256)
void vq_gather(const float* __restrict__ cb, const int* __restrict__ amin,
               float* __restrict__ emb, float* __restrict__ recon) {
  const size_t i = (size_t)blockIdx.x * 256 + threadIdx.x;
  if (i >= BLTOT) return;
  const int m = (int)(i >> 9);
  const int l = (int)(i & (NL - 1));
  const float v = cb[(size_t)l * NL + amin[m]];
  emb[i] = v;
  recon[i] = v;
}

// ---------------- orchestration ----------------
extern "C" void kernel_launch(void* const* d_in, const int* in_sizes, int n_in,
                              void* d_out, int out_size, void* d_ws, size_t ws_size,
                              hipStream_t stream) {
  const float* x    = (const float*)d_in[0];
  const float* cb   = (const float*)d_in[1];
  const float* ew1  = (const float*)d_in[2];
  const float* eb1  = (const float*)d_in[3];
  const float* ew2  = (const float*)d_in[4];
  const float* eb2  = (const float*)d_in[5];
  const float* dw1  = (const float*)d_in[6];
  const float* db1  = (const float*)d_in[7];
  const float* dw2  = (const float*)d_in[8];
  const float* db2  = (const float*)d_in[9];
  const float* eg   = (const float*)d_in[10];
  const float* ebn  = (const float*)d_in[11];
  const float* dg   = (const float*)d_in[12];
  const float* dbn  = (const float*)d_in[13];

  // output layout: y | h | emb | argmin(f32) | recon
  float* out       = (float*)d_out;
  float* out_y     = out;
  float* out_h     = out + BLTOT;
  float* out_emb   = out + 2 * BLTOT;
  float* out_arg   = out + 3 * BLTOT;
  float* out_recon = out_arg + NB;
  float* t3        = out_y;                 // f32 scratch until final GEMM

  // workspace layout
  char* w = (char*)d_ws;
  float* t0 = (float*)w;                    w += BLTOT * 4;       // f32 temp (h chain)
  char* R = w;                              w += BLTOT * 4;       // 128MB aliased region
  unsigned short* xbf    = (unsigned short*)R;                    // [0,64MB): bf16 x
  unsigned short* t2bf   = (unsigned short*)(R + BLTOT * 2);      // [64,128MB): LN->GEMM
  float*          scores = (float*)R;                             // whole R at VQ time
  unsigned short* t1bf   = (unsigned short*)w; w += BLTOT * 2;    // GELU out / h_bf
  unsigned short* wtE1 = (unsigned short*)w; w += 3 * WSZ * 2;
  unsigned short* wtE2 = (unsigned short*)w; w += 3 * WSZ * 2;
  unsigned short* wtD1 = (unsigned short*)w; w += 3 * WSZ * 2;
  unsigned short* wtD2 = (unsigned short*)w; w += 3 * WSZ * 2;
  unsigned short* cbt  = (unsigned short*)w; w += WSZ * 2;
  int*   amin = (int*)w;                    w += (size_t)NB * 4;
  float* cbn  = (float*)w;                  w += NL * 4;
  float* psum = (float*)w;                  w += NGROUP * 32 * 4;
  float* psq  = (float*)w;                  w += NGROUP * 32 * 4;
  float* mu   = (float*)w;                  w += NGROUP * 4;
  float* rstd = (float*)w;

  dim3 ggrid(NL / BN, NB / BM);   // 4 x 512
  dim3 gblk(256);

  auto GEMM_GELU = [&](const unsigned short* A, const unsigned short* Wt,
                       const float* b, unsigned short* O) {
    gemm512<0><<<ggrid, gblk, 0, stream>>>(A, Wt, b, nullptr, nullptr, O);
  };
  auto GEMM_BIAS = [&](const unsigned short* A, const unsigned short* Wt,
                       const float* b, const float* r, float* O) {
    gemm512<1><<<ggrid, gblk, 0, stream>>>(A, Wt, b, r, O, nullptr);
  };
  auto RLN = [&](const float* in, const float* g, const float* b,
                 float* Of, unsigned short* Ob) {
    rln_partial<<<dim3(NGROUP, 32), 256, 0, stream>>>(in, psum, psq);
    rln_finalize<<<1, 64, 0, stream>>>(psum, psq, mu, rstd);
    rln_apply<<<(unsigned)(BLTOT / 4 / 256), 256, 0, stream>>>(in, g, b, mu, rstd, Of, Ob);
  };

  // -------- one-time converts (bf16, weights pre-transposed) --------
  conv_plain<<<(unsigned)(BLTOT / 4 / 256), 256, 0, stream>>>(x, xbf, BLTOT);
  conv_transpose<<<(unsigned)(3 * WSZ / 256), 256, 0, stream>>>(ew1, wtE1, 3);
  conv_transpose<<<(unsigned)(3 * WSZ / 256), 256, 0, stream>>>(ew2, wtE2, 3);
  conv_transpose<<<(unsigned)(3 * WSZ / 256), 256, 0, stream>>>(dw1, wtD1, 3);
  conv_transpose<<<(unsigned)(3 * WSZ / 256), 256, 0, stream>>>(dw2, wtD2, 3);
  conv_transpose<<<(unsigned)(WSZ / 256), 256, 0, stream>>>(cb, cbt, 1);
  cb_norm<<<2, 256, 0, stream>>>(cb, cbn);

  // -------- encoder --------
  GEMM_GELU(xbf, wtE1 + 0 * WSZ, eb1 + 0 * NL, t1bf);
  GEMM_BIAS(t1bf, wtE2 + 0 * WSZ, eb2 + 0 * NL, nullptr, t0);          // h0
  RLN(t0, eg + 0 * (size_t)GRP, ebn + 0 * (size_t)GRP, nullptr, t2bf);
  GEMM_GELU(t2bf, wtE1 + 1 * WSZ, eb1 + 1 * NL, t1bf);
  GEMM_BIAS(t1bf, wtE2 + 1 * WSZ, eb2 + 1 * NL, t0, t3);               // h1 -> t3
  RLN(t3, eg + 1 * (size_t)GRP, ebn + 1 * (size_t)GRP, nullptr, t2bf);
  GEMM_GELU(t2bf, wtE1 + 2 * WSZ, eb1 + 2 * NL, t1bf);
  GEMM_BIAS(t1bf, wtE2 + 2 * WSZ, eb2 + 2 * NL, t3, t0);               // h1 + mlp
  RLN(t0, eg + 2 * (size_t)GRP, ebn + 2 * (size_t)GRP, out_h, t1bf);   // h + bf16 shadow

  // -------- VQ --------
  gemm512<2><<<ggrid, gblk, 0, stream>>>(t1bf, cbt, cbn, nullptr, scores, nullptr);
  argmin_rows<<<NB / 8, 256, 0, stream>>>(scores, amin, out_arg);
  vq_gather<<<(unsigned)(BLTOT / 256), 256, 0, stream>>>(cb, amin, out_emb, out_recon);

  // -------- decoder (q = out_emb) --------
  RLN(out_emb, dg + 0 * (size_t)GRP, dbn + 0 * (size_t)GRP, nullptr, t2bf);
  GEMM_GELU(t2bf, wtD1 + 1 * WSZ, db1 + 1 * NL, t1bf);
  GEMM_BIAS(t1bf, wtD2 + 1 * WSZ, db2 + 1 * NL, out_emb, t0);          // y1
  RLN(t0, dg + 1 * (size_t)GRP, dbn + 1 * (size_t)GRP, nullptr, t2bf);
  GEMM_GELU(t2bf, wtD1 + 2 * WSZ, db1 + 2 * NL, t1bf);
  GEMM_BIAS(t1bf, wtD2 + 2 * WSZ, db2 + 2 * NL, t0, t3);               // y2 -> out_y scratch
  RLN(t3, dg + 2 * (size_t)GRP, dbn + 2 * (size_t)GRP, nullptr, t2bf); // y3
  GEMM_GELU(t2bf, wtD1 + 0 * WSZ, db1 + 0 * NL, t1bf);
  GEMM_BIAS(t1bf, wtD2 + 0 * WSZ, db2 + 0 * NL, nullptr, out_y);       // final y
}
